// SolvGNNV7_37778532335678
// MI455X (gfx1250) — matrix operand order repaired and checked by reading, not compiled, not run on hardware
//
#include <hip/hip_runtime.h>
#include <math.h>

// ---------------- CDNA5 WMMA types ----------------
typedef __bf16 v16bf __attribute__((ext_vector_type(16)));
typedef float  v8f   __attribute__((ext_vector_type(8)));

union FragB16 { v16bf v; uint4 q[2]; unsigned short u[16]; };

__device__ __forceinline__ unsigned short f2bf(float f) {
  unsigned int u = __float_as_uint(f);
  u += 0x7FFFu + ((u >> 16) & 1u);          // round-to-nearest-even
  return (unsigned short)(u >> 16);
}
__device__ __forceinline__ float bf2f(unsigned short h) {
  return __uint_as_float(((unsigned int)h) << 16);
}

// ---------------- workspace layout (bf16/ushort units) ----------------
#define OFF_WIN  0u                         // 96*256
#define OFF_G1   24576u                     // 5*256*256
#define OFF_G2   352256u                    // 5*256*256
#define OFF_Q    679936u                    // 256*256
#define OFF_K    745472u                    // 256*256
#define OFF_M1   811008u                    // 288*1024
#define OFF_M2   1105920u                   // 1024*512
#define OFF_Z    1630208u                   // 4096*288
#define OFF_Z1   2809856u                   // 4096*1024
#define OFF_Z2   7004160u                   // 4096*512

// ---------------- pack fp32 weight -> bf16 in B-fragment order ----------------
// B layout packed here (contiguous k, same order as A): lane holds n = lane&15, K = e + (lane>>4)*16
// packed[((nt*KK + kk)*32 + lane)*16 + e] = W[kk*32 + (lane>>4)*16 + e][nt*16 + (lane&15)]
__global__ __launch_bounds__(256) void pack_w_kernel(const float* __restrict__ W,
                                                     unsigned short* __restrict__ out,
                                                     int Korig, int KK, int N) {
  size_t idx = (size_t)blockIdx.x * 256 + threadIdx.x;
  size_t total = (size_t)KK * 32 * N;
  if (idx >= total) return;
  int e    = (int)(idx & 15);
  size_t t = idx >> 4;
  int lane = (int)(t & 31); t >>= 5;
  int kk   = (int)(t % (size_t)KK);
  int nt   = (int)(t / (size_t)KK);
  int k = kk * 32 + (lane >> 4) * 16 + e;
  int n = nt * 16 + (lane & 15);
  float v = (k < Korig) ? W[(size_t)k * N + n] : 0.f;
  out[idx] = f2bf(v);
}

// ---------------- one 16x16 D tile via WMMA over K ----------------
__device__ __forceinline__ v8f wmma_tile(const unsigned short* shA, int srow,
                                         const unsigned short* Wp, int KK,
                                         int mt, int nt, int lane) {
  v8f acc = {0.f, 0.f, 0.f, 0.f, 0.f, 0.f, 0.f, 0.f};
  const int m    = mt * 16 + (lane & 15);
  const int half = lane >> 4;
  const unsigned short* arow = shA + (size_t)m * srow + half * 16;
  const uint4* wp = (const uint4*)(Wp + (((size_t)nt * KK) * 32 + lane) * 16);
  for (int kk = 0; kk < KK; ++kk) {
    FragB16 a, b;
    // 16-bit A layout (16x32): K={0..15}+half*16 (contiguous, matches packed B)
    a.q[0] = *(const uint4*)(arow);
    a.q[1] = *(const uint4*)(arow + 8);
    b.q[0] = wp[0];
    b.q[1] = wp[1];
    __builtin_prefetch(wp + 64, 0, 0);      // next K-chunk of weights
    acc = __builtin_amdgcn_wmma_f32_16x16x32_bf16(false, a.v, false, b.v,
                                                  (short)0, acc, false, false);
    arow += 32;
    wp   += 64;                              // 32 lanes * 16 bf16 = 64 uint4
  }
  return acc;
}

// [32 x K] @ [K x 256] -> shC fp32 [32][256]; 8 waves x 4 tiles
__device__ __forceinline__ void mm_to_f32(const unsigned short* shA, int srow, int KK,
                                          const unsigned short* Wp, float* shC,
                                          int wv, int lane) {
  for (int j = 0; j < 4; ++j) {
    int mt = j & 1, nt = wv * 2 + (j >> 1);
    v8f acc = wmma_tile(shA, srow, Wp, KK, mt, nt, lane);
    int half = lane >> 4;
    float* crow = shC + (size_t)(mt * 16 + half * 8) * 256 + nt * 16 + (lane & 15);
#pragma unroll
    for (int r = 0; r < 8; ++r) crow[(size_t)r * 256] = acc[r];
  }
}

// deterministic edge gather + inorm*x + bias [+relu]
__device__ __forceinline__ void gather_finalize(const float* shM, float* shO,
                                                const short* adj, const unsigned char* cnt,
                                                const float* inorm, const float* __restrict__ bias,
                                                bool relu, int tid) {
  for (int i = tid; i < 32 * 256; i += 256) {
    int d = i >> 8, c = i & 255;
    float acc = 0.f;
    int n = cnt[d];
    for (int e = 0; e < n; ++e) acc += shM[(size_t)adj[d * 64 + e] * 256 + c];
    float v = acc * inorm[d] + bias[c];
    if (relu) v = fmaxf(v, 0.f);
    shO[i] = v;
  }
}

__device__ __forceinline__ void store_h_bf16(const float* shI, unsigned short* shH,
                                             const float* onorm, bool relu, int tid) {
  for (int i = tid; i < 32 * 256; i += 256) {
    float v = shI[i];
    if (relu) v = fmaxf(v, 0.f);
    shH[i] = f2bf(v * onorm[i >> 8]);
  }
}

// ---------------- per-graph GNN kernel: 1 block = 1 graph ----------------
__global__ __launch_bounds__(256) void solvgnn_graph_kernel(
    const float* __restrict__ x, const float* __restrict__ addf,
    const int* __restrict__ src, const int* __restrict__ dst,
    const unsigned short* __restrict__ ws,
    const float* __restrict__ b_in, const float* __restrict__ b_g1,
    const float* __restrict__ b_g2, const float* __restrict__ bq,
    const float* __restrict__ bk, const float* __restrict__ Wv,
    const float* __restrict__ bv, unsigned short* __restrict__ z_out) {
  extern __shared__ char smem[];
  float* sh_f            = (float*)smem;                       // 32KB: matmul out
  float* sh_agg          = (float*)(smem + 32768);             // 32KB: fp32 h
  unsigned short* sh_h   = (unsigned short*)(smem + 65536);    // 16KB: bf16 A operand

  __shared__ float s_S[1024];
  __shared__ float s_v[32], s_w[32], s_onorm[32], s_inorm[32];
  __shared__ int   s_odeg[32], s_ideg[32], s_src[64], s_dst[64];
  __shared__ short s_adj[32 * 64];
  __shared__ unsigned char s_cnt[32];

  const int g = blockIdx.x, tid = threadIdx.x;
  const int wv = tid >> 5, lane = tid & 31;

  if (tid < 64) { s_src[tid] = src[(size_t)g * 64 + tid]; s_dst[tid] = dst[(size_t)g * 64 + tid]; }
  if (tid < 32) { s_odeg[tid] = 0; s_ideg[tid] = 0; }
  __syncthreads();
  if (tid < 64) { atomicAdd(&s_odeg[s_src[tid]], 1); atomicAdd(&s_ideg[s_dst[tid]], 1); }
  __syncthreads();
  if (tid < 32) {
    s_onorm[tid] = rsqrtf(fmaxf((float)s_odeg[tid], 1.f));
    s_inorm[tid] = rsqrtf(fmaxf((float)s_ideg[tid], 1.f));
    int cnt = 0;
    for (int e = 0; e < 64; ++e)
      if (s_dst[e] == tid) s_adj[tid * 64 + cnt++] = (short)s_src[e];
    s_cnt[tid] = (unsigned char)cnt;
  }
  __syncthreads();

  // ---- first conv: x (pad 74->96) scaled by onorm ----
  for (int i = tid; i < 32 * 96; i += 256) {
    int n = i / 96, c = i - n * 96;
    float v = (c < 74) ? x[((size_t)g * 32 + n) * 74 + c] : 0.f;
    sh_h[n * 96 + c] = f2bf(v * s_onorm[n]);
  }
  __syncthreads();
  mm_to_f32(sh_h, 96, 3, ws + OFF_WIN, sh_f, wv, lane);
  __syncthreads();
  gather_finalize(sh_f, sh_agg, s_adj, s_cnt, s_inorm, b_in, false, tid);
  __syncthreads();

  // ---- 5 GCR blocks ----
  for (int L = 0; L < 5; ++L) {
    store_h_bf16(sh_agg, sh_h, s_onorm, false, tid);
    __syncthreads();
    mm_to_f32(sh_h, 256, 8, ws + OFF_G1 + (size_t)L * 65536, sh_f, wv, lane);
    __syncthreads();
    gather_finalize(sh_f, sh_agg, s_adj, s_cnt, s_inorm, b_g1 + L * 256, false, tid);
    __syncthreads();
    store_h_bf16(sh_agg, sh_h, s_onorm, true, tid);     // relu before second conv
    __syncthreads();
    mm_to_f32(sh_h, 256, 8, ws + OFF_G2 + (size_t)L * 65536, sh_f, wv, lane);
    __syncthreads();
    gather_finalize(sh_f, sh_agg, s_adj, s_cnt, s_inorm, b_g2 + L * 256, true, tid); // h=relu(t)
    __syncthreads();
  }

  // ---- attention pooling: sh_agg = final h (fp32, relu'd) ----
  for (int i = tid; i < 8192; i += 256) sh_h[i] = f2bf(sh_agg[i]);   // unscaled h
  __syncthreads();
  mm_to_f32(sh_h, 256, 8, ws + OFF_Q, sh_f, wv, lane);               // q -> sh_f
  __syncthreads();
  v8f kacc[4];
  for (int j = 0; j < 4; ++j) {                                      // k -> regs
    int mt = j & 1, nt = wv * 2 + (j >> 1);
    kacc[j] = wmma_tile(sh_h, 256, ws + OFF_K, 8, mt, nt, lane);
  }
  for (int i = tid; i < 8192; i += 256) sh_f[i] += bq[i & 255];      // q bias
  __syncthreads();
  for (int j = 0; j < 4; ++j) {                                      // k (+bias) -> sh_h bf16
    int mt = j & 1, nt = wv * 2 + (j >> 1);
    int half = lane >> 4, col = nt * 16 + (lane & 15);
    unsigned short* kr = sh_h + (size_t)(mt * 16 + half * 8) * 256 + col;
#pragma unroll
    for (int r = 0; r < 8; ++r) kr[(size_t)r * 256] = f2bf(kacc[j][r] + bk[col]);
  }
  if (tid < 32) {                                                    // v = h@Wv + bv
    float acc = bv[0];
    for (int c = 0; c < 256; ++c) acc += sh_agg[(size_t)tid * 256 + c] * Wv[c];
    s_v[tid] = acc;
  }
  __syncthreads();
  for (int i = tid; i < 1024; i += 256) {                            // S = q k^T
    int r = i >> 5, c2 = i & 31;
    float s = 0.f;
    for (int c = 0; c < 256; ++c) s += sh_f[(size_t)r * 256 + c] * bf2f(sh_h[(size_t)c2 * 256 + c]);
    s_S[i] = s;
  }
  __syncthreads();
  if (tid < 32) {                                                    // softmax row + @v
    float mx = -1e30f;
    for (int j = 0; j < 32; ++j) mx = fmaxf(mx, s_S[tid * 32 + j]);
    float se = 0.f, dot = 0.f;
    for (int j = 0; j < 32; ++j) {
      float e = __expf(s_S[tid * 32 + j] - mx);
      se += e; dot += e * s_v[j];
    }
    s_w[tid] = dot / se;
  }
  __syncthreads();
  {                                                                  // feats = h^T @ w -> z row
    float f = 0.f;
    for (int i = 0; i < 32; ++i) f += sh_agg[(size_t)i * 256 + tid] * s_w[i];
    z_out[(size_t)g * 288 + tid] = f2bf(f);
    if (tid < 32) {
      unsigned short val = 0;
      if (tid == 0)      val = f2bf(addf[(size_t)g * 2]);
      else if (tid == 1) val = f2bf(addf[(size_t)g * 2 + 1]);
      z_out[(size_t)g * 288 + 256 + tid] = val;
    }
  }
}

// ---------------- batched MLP GEMM: leaky_relu(A @ W + b) -> bf16 ----------------
// A: [4096 x K] bf16 row-major (K = KK*32), Wp packed, 32 rows per block
__global__ __launch_bounds__(256) void gemm_bf16_kernel(const unsigned short* __restrict__ A,
                                                        const unsigned short* __restrict__ Wp,
                                                        const float* __restrict__ bias,
                                                        unsigned short* __restrict__ out,
                                                        int N, int KK) {
  extern __shared__ char smem[];
  unsigned short* shA = (unsigned short*)smem;       // 32 * K bf16
  const int K = KK * 32;
  const int tid = threadIdx.x, wv = tid >> 5, lane = tid & 31;

  const uint4* gsrc = (const uint4*)(A + (size_t)blockIdx.x * 32 * K);
  uint4* ldst = (uint4*)shA;
  const int n16 = (32 * K) >> 3;                     // uint4 = 8 bf16
  for (int i = tid; i < n16; i += 256) ldst[i] = gsrc[i];
  __syncthreads();

  const int ntiles = 2 * (N >> 4);
  for (int t = wv; t < ntiles; t += 8) {
    int mt = t & 1, nt = t >> 1;
    v8f acc = wmma_tile(shA, K, Wp, KK, mt, nt, lane);
    int half = lane >> 4, col = nt * 16 + (lane & 15);
    size_t rowbase = (size_t)blockIdx.x * 32 + mt * 16 + half * 8;
#pragma unroll
    for (int r = 0; r < 8; ++r) {
      float v = acc[r] + bias[col];
      v = (v > 0.f) ? v : 0.01f * v;                 // leaky_relu
      out[(rowbase + r) * N + col] = f2bf(v);
    }
  }
}

// ---------------- final 512 -> 1 layer: one wave per row ----------------
__global__ __launch_bounds__(256) void final_dot_kernel(const unsigned short* __restrict__ z2,
                                                        const float* __restrict__ W3,
                                                        const float* __restrict__ b3,
                                                        float* __restrict__ out) {
  int row = blockIdx.x * 8 + (threadIdx.x >> 5);
  int lane = threadIdx.x & 31;
  float acc = 0.f;
  for (int c = lane; c < 512; c += 32) acc += bf2f(z2[(size_t)row * 512 + c]) * W3[c];
#pragma unroll
  for (int off = 16; off; off >>= 1) acc += __shfl_xor(acc, off, 32);
  if (lane == 0) out[row] = acc + b3[0];
}

// ---------------- host launcher ----------------
extern "C" void kernel_launch(void* const* d_in, const int* in_sizes, int n_in,
                              void* d_out, int out_size, void* d_ws, size_t ws_size,
                              hipStream_t stream) {
  const float* x    = (const float*)d_in[0];
  const float* addf = (const float*)d_in[1];
  const int*   src  = (const int*)d_in[2];
  const int*   dst  = (const int*)d_in[3];
  const float* W_in = (const float*)d_in[4];
  const float* b_in = (const float*)d_in[5];
  const float* W_g1 = (const float*)d_in[6];
  const float* b_g1 = (const float*)d_in[7];
  const float* W_g2 = (const float*)d_in[8];
  const float* b_g2 = (const float*)d_in[9];
  const float* Wq   = (const float*)d_in[10];
  const float* bq   = (const float*)d_in[11];
  const float* Wk   = (const float*)d_in[12];
  const float* bk   = (const float*)d_in[13];
  const float* Wv   = (const float*)d_in[14];
  const float* bv   = (const float*)d_in[15];
  const float* W_m1 = (const float*)d_in[16];
  const float* b_m1 = (const float*)d_in[17];
  const float* W_m2 = (const float*)d_in[18];
  const float* b_m2 = (const float*)d_in[19];
  const float* W_m3 = (const float*)d_in[20];
  const float* b_m3 = (const float*)d_in[21];
  float* out = (float*)d_out;
  unsigned short* ws = (unsigned short*)d_ws;

  // --- pack all GEMM weights to bf16 fragment layout ---
  pack_w_kernel<<<96, 256, 0, stream>>>(W_in, ws + OFF_WIN, 74, 3, 256);
  for (int i = 0; i < 5; ++i) {
    pack_w_kernel<<<256, 256, 0, stream>>>(W_g1 + (size_t)i * 65536, ws + OFF_G1 + (size_t)i * 65536, 256, 8, 256);
    pack_w_kernel<<<256, 256, 0, stream>>>(W_g2 + (size_t)i * 65536, ws + OFF_G2 + (size_t)i * 65536, 256, 8, 256);
  }
  pack_w_kernel<<<256, 256, 0, stream>>>(Wq, ws + OFF_Q, 256, 8, 256);
  pack_w_kernel<<<256, 256, 0, stream>>>(Wk, ws + OFF_K, 256, 8, 256);
  pack_w_kernel<<<1152, 256, 0, stream>>>(W_m1, ws + OFF_M1, 258, 9, 1024);
  pack_w_kernel<<<2048, 256, 0, stream>>>(W_m2, ws + OFF_M2, 1024, 32, 512);

  // --- per-graph GNN: conv stack + attention pooling -> z rows (bf16, K=288) ---
  solvgnn_graph_kernel<<<4096, 256, 81920, stream>>>(
      x, addf, src, dst, ws, b_in, b_g1, b_g2, bq, bk, Wv, bv, ws + OFF_Z);

  // --- MLP ---
  gemm_bf16_kernel<<<128, 256, 32 * 288 * 2, stream>>>(ws + OFF_Z,  ws + OFF_M1, b_m1, ws + OFF_Z1, 1024, 9);
  gemm_bf16_kernel<<<128, 256, 32 * 1024 * 2, stream>>>(ws + OFF_Z1, ws + OFF_M2, b_m2, ws + OFF_Z2, 512, 32);
  final_dot_kernel<<<512, 256, 0, stream>>>(ws + OFF_Z2, W_m3, b_m3, out);
}
